// VFcont_65738769432779
// MI455X (gfx1250) — matrix operand order repaired and checked by reading, not compiled, not run on hardware
//
#include <hip/hip_runtime.h>
#include <stdint.h>

typedef __bf16 bf16;
typedef __attribute__((ext_vector_type(16))) __bf16 v16bf;
typedef __attribute__((ext_vector_type(8)))  __bf16 v8bf;
typedef __attribute__((ext_vector_type(8)))  float  v8f;
typedef int v4i __attribute__((vector_size(16)));   // matches builtin signature

#define BATCH 256
#define N0    512
#define N1    2048
#define NIN   4096
#define TSTEPS 60

// LDS staging: 64 rows x 32 K-elements per stage, rows padded to 48 elements
// (96B stride: keeps 16B alignment for b128 while staggering bank groups).
#define LDSROW      48
#define STAGE_ELEMS (64 * LDSROW)
#define STAGES      4

#if __has_builtin(__builtin_amdgcn_global_load_async_to_lds_b128)
#define ASYNC_OK 1
#else
#define ASYNC_OK 0
#endif

// ---- CDNA5 async global->LDS copy (ASYNCcnt-tracked), with safe fallback ----
__device__ __forceinline__ void copy16(const bf16* g, bf16* l) {
#if ASYNC_OK
  typedef __attribute__((address_space(1))) v4i gv4i;
  typedef __attribute__((address_space(3))) v4i lv4i;
  __builtin_amdgcn_global_load_async_to_lds_b128(
      (gv4i*)(uintptr_t)g,
      (lv4i*)(uint32_t)(uintptr_t)l,
      0, 0);
#else
  *(v8bf*)l = *(const v8bf*)g;   // sync fallback: global_load_b128 + ds_store_b128
#endif
}

template <int N>
__device__ __forceinline__ void wait_async() {
#if ASYNC_OK
#if __has_builtin(__builtin_amdgcn_s_wait_asynccnt)
  __builtin_amdgcn_s_wait_asynccnt(N);
#else
  asm volatile("s_wait_asynccnt %0" ::"i"(N) : "memory");
#endif
#endif
}

__device__ __forceinline__ v16bf cat8(v8bf lo, v8bf hi) {
  v16bf r;
#pragma unroll
  for (int e = 0; e < 8; ++e) { r[e] = lo[e]; r[e + 8] = hi[e]; }
  return r;
}

// Stage one 64x32 A slice and one 64x32 B slice into LDS.
// 256 chunks of 16B split over 128 threads (2 per thread per operand);
// each wave issues exactly 4 async ops per stage.
__device__ __forceinline__ void issue_stage(const bf16* __restrict__ A, int lda,
                                            const bf16* __restrict__ Bn, int ldb,
                                            bf16* As, bf16* Bs, int kk, int tid) {
#pragma unroll
  for (int h = 0; h < 2; ++h) {
    const int c   = tid + h * 128;
    const int row = c >> 2;
    const int ko  = (c & 3) * 8;
    copy16(A  + (size_t)row * lda + kk + ko, As + row * LDSROW + ko);
    copy16(Bn + (size_t)row * ldb + kk + ko, Bs + row * LDSROW + ko);
  }
}

// Block-cooperative 64x64 macro-tile of C = A(MxK) x B^T, Bn stored N-major
// (Bn[n][k]). 4 waves, each owning a 32x32 quadrant; 4-stage async pipeline.
// Fragment layouts per CDNA5 ISA (wave32):
//  A 16x32 bf16: lane r=lane&15 = row M; half=lane>>4 -> K chunks
//                [8*half,+8) and [16+8*half,+8) of the slice    -> 2x ds_load_b128
//  B 32x16 bf16: lane r = column N; half -> K chunks [8*half,+8) and [16+8*half,+8) -> 2x ds_load_b128
template <int K>
__device__ __forceinline__ void gemm_block64(const bf16* __restrict__ A, int lda,
                                             const bf16* __restrict__ Bn, int ldb,
                                             bf16* As, bf16* Bs, int tid,
                                             v8f acc[2][2]) {
  const int lane = tid & 31;
  const int w    = tid >> 5;
  const int r    = lane & 15, half = lane >> 4;
  const int wm   = (w >> 1) * 32;
  const int wn   = (w & 1) * 32;
  constexpr int NIT = K / 32;   // >= STAGES for all call sites (16/64/128)

#pragma unroll
  for (int s = 0; s < STAGES; ++s)
    issue_stage(A, lda, Bn, ldb, As + s * STAGE_ELEMS, Bs + s * STAGE_ELEMS, s * 32, tid);

  for (int it = 0; it < NIT; ++it) {
    // Oldest stage is done when ASYNCcnt <= 4 * (#newer stages in flight).
    const int rem   = NIT - 1 - it;
    const int newer = rem < (STAGES - 1) ? rem : (STAGES - 1);
    switch (newer) {            // wave-uniform; only varies in the tail
      case 3:  wait_async<12>(); break;
      case 2:  wait_async<8>();  break;
      case 1:  wait_async<4>();  break;
      default: wait_async<0>();  break;
    }
    __syncthreads();            // all waves' copies for this stage are visible

    const bf16* as = As + (it & (STAGES - 1)) * STAGE_ELEMS;
    const bf16* bs = Bs + (it & (STAGES - 1)) * STAGE_ELEMS;
    v16bf a[2], b[2];
#pragma unroll
    for (int i = 0; i < 2; ++i) {
      const bf16* ap = as + (wm + i * 16 + r) * LDSROW + half * 8;
      a[i] = cat8(*(const v8bf*)ap, *(const v8bf*)(ap + 16));
      const bf16* bp = bs + (wn + i * 16 + r) * LDSROW + half * 8;
      b[i] = cat8(*(const v8bf*)bp, *(const v8bf*)(bp + 16));
    }
#pragma unroll
    for (int mi = 0; mi < 2; ++mi)
#pragma unroll
      for (int ni = 0; ni < 2; ++ni)
        acc[mi][ni] = __builtin_amdgcn_wmma_f32_16x16x32_bf16(
            false, a[mi], false, b[ni], (short)0, acc[mi][ni], false, false);

    __syncthreads();            // everyone done reading this buffer
    if (it + STAGES < NIT)      // refill the buffer just consumed
      issue_stage(A, lda, Bn, ldb,
                  As + (it & (STAGES - 1)) * STAGE_ELEMS,
                  Bs + (it & (STAGES - 1)) * STAGE_ELEMS,
                  (it + STAGES) * 32, tid);
  }
}

__global__ void cvt_bf16_kernel(const float* __restrict__ src, bf16* __restrict__ dst,
                                int n, int doclip) {
  int i = blockIdx.x * blockDim.x + threadIdx.x;
  if (i < n) {
    float v = src[i];
    if (doclip) v = fminf(fmaxf(v, 0.0f), 1.0f);
    dst[i] = (bf16)v;
  }
}

__global__ void init_state_kernel(const float* __restrict__ src, float* __restrict__ dstf,
                                  bf16* __restrict__ dstb, int n) {
  int i = blockIdx.x * blockDim.x + threadIdx.x;
  if (i < n) {
    float v = src[i];
    dstf[i] = v;
    dstb[i] = (bf16)fminf(fmaxf(v, 0.0f), 1.0f);  // rho(state)
  }
}

// inp1 = rho(data) @ w2^T + b2 : M=BATCH, N=N1, K=NIN. 128 blocks of 64x64.
__global__ __launch_bounds__(128) void inp1_kernel(const bf16* __restrict__ rd,
                                                   const bf16* __restrict__ w2b,
                                                   const float* __restrict__ b2,
                                                   float* __restrict__ inp1) {
  __shared__ bf16 As[STAGES * STAGE_ELEMS];
  __shared__ bf16 Bs[STAGES * STAGE_ELEMS];
  const int tid  = threadIdx.x;
  const int lane = tid & 31;
  const int w    = tid >> 5;
  const int r    = lane & 15, half = lane >> 4;
  const int tm   = (blockIdx.x >> 5) * 64;   // 4 M-blocks
  const int tn   = (blockIdx.x & 31) * 64;   // 32 N-blocks
  v8f acc[2][2] = {};
  gemm_block64<NIN>(rd + (size_t)tm * NIN, NIN, w2b + (size_t)tn * NIN, NIN, As, Bs, tid, acc);
  const int wm = (w >> 1) * 32, wn = (w & 1) * 32;
#pragma unroll
  for (int mi = 0; mi < 2; ++mi)
#pragma unroll
    for (int ni = 0; ni < 2; ++ni) {
      const int col = tn + wn + ni * 16 + r;
      const float bias = b2[col];
#pragma unroll
      for (int j = 0; j < 8; ++j) {
        const int row = tm + wm + mi * 16 + half * 8 + j;
        inp1[(size_t)row * N1 + col] = acc[mi][ni][j] + bias;
      }
    }
}

// One Euler step (dt=0.5):
//  blocks [0,32):    s0' = clip(0.5*s0 + 0.5*(s1 @ w0^T + b0))   (K=N1)
//  blocks [32,160):  s1' = clip(0.5*s1 + 0.5*(inp1 + s0 @ w1^T)) (K=N0)
__global__ __launch_bounds__(128) void step_kernel(
    const float* __restrict__ s0f_in, const float* __restrict__ s1f_in,
    const bf16* __restrict__ s0b_in, const bf16* __restrict__ s1b_in,
    const bf16* __restrict__ w0b, const bf16* __restrict__ w1b,
    const float* __restrict__ b0, const float* __restrict__ inp1,
    float* __restrict__ s0f_out, float* __restrict__ s1f_out,
    bf16* __restrict__ s0b_out, bf16* __restrict__ s1b_out) {
  __shared__ bf16 As[STAGES * STAGE_ELEMS];
  __shared__ bf16 Bs[STAGES * STAGE_ELEMS];
  const int tid  = threadIdx.x;
  const int lane = tid & 31;
  const int w    = tid >> 5;
  const int r    = lane & 15, half = lane >> 4;
  const int wm   = (w >> 1) * 32, wn = (w & 1) * 32;
  v8f acc[2][2] = {};
  if (blockIdx.x < 32) {
    const int tm = (blockIdx.x >> 3) * 64;   // 4 M-blocks
    const int tn = (blockIdx.x & 7) * 64;    // 8 N-blocks over N0
    gemm_block64<N1>(s1b_in + (size_t)tm * N1, N1, w0b + (size_t)tn * N1, N1, As, Bs, tid, acc);
#pragma unroll
    for (int mi = 0; mi < 2; ++mi)
#pragma unroll
      for (int ni = 0; ni < 2; ++ni) {
        const int col = tn + wn + ni * 16 + r;
        const float bias = b0[col];
#pragma unroll
        for (int j = 0; j < 8; ++j) {
          const int row = tm + wm + mi * 16 + half * 8 + j;
          const size_t idx = (size_t)row * N0 + col;
          float v = 0.5f * s0f_in[idx] + 0.5f * (acc[mi][ni][j] + bias);
          v = fminf(fmaxf(v, 0.0f), 1.0f);
          s0f_out[idx] = v;
          s0b_out[idx] = (bf16)v;            // clipped state: rho(s') == s'
        }
      }
  } else {
    const int b2i = blockIdx.x - 32;
    const int tm = (b2i >> 5) * 64;          // 4 M-blocks
    const int tn = (b2i & 31) * 64;          // 32 N-blocks over N1
    gemm_block64<N0>(s0b_in + (size_t)tm * N0, N0, w1b + (size_t)tn * N0, N0, As, Bs, tid, acc);
#pragma unroll
    for (int mi = 0; mi < 2; ++mi)
#pragma unroll
      for (int ni = 0; ni < 2; ++ni) {
        const int col = tn + wn + ni * 16 + r;
#pragma unroll
        for (int j = 0; j < 8; ++j) {
          const int row = tm + wm + mi * 16 + half * 8 + j;
          const size_t idx = (size_t)row * N1 + col;
          float v = 0.5f * s1f_in[idx] + 0.5f * (inp1[idx] + acc[mi][ni][j]);
          v = fminf(fmaxf(v, 0.0f), 1.0f);
          s1f_out[idx] = v;
          s1b_out[idx] = (bf16)v;
        }
      }
  }
}

extern "C" void kernel_launch(void* const* d_in, const int* in_sizes, int n_in,
                              void* d_out, int out_size, void* d_ws, size_t ws_size,
                              hipStream_t stream) {
  (void)in_sizes; (void)n_in; (void)out_size; (void)ws_size;
  const float* data = (const float*)d_in[0];
  const float* s0in = (const float*)d_in[1];
  const float* s1in = (const float*)d_in[2];
  const float* w0   = (const float*)d_in[3];
  const float* b0   = (const float*)d_in[4];
  const float* w1   = (const float*)d_in[5];
  const float* w2   = (const float*)d_in[6];
  const float* b2   = (const float*)d_in[7];

  char* ws = (char*)d_ws;
  size_t off = 0;
  auto alloc = [&](size_t bytes) -> char* {
    char* p = ws + off;
    off += (bytes + 255) & ~(size_t)255;
    return p;
  };
  bf16*  rd    = (bf16*)alloc((size_t)BATCH * NIN * 2);
  bf16*  w0b   = (bf16*)alloc((size_t)N0 * N1 * 2);
  bf16*  w1b   = (bf16*)alloc((size_t)N1 * N0 * 2);
  bf16*  w2b   = (bf16*)alloc((size_t)N1 * NIN * 2);
  float* inp1  = (float*)alloc((size_t)BATCH * N1 * 4);
  float* s0f[2] = {(float*)alloc((size_t)BATCH * N0 * 4), (float*)alloc((size_t)BATCH * N0 * 4)};
  float* s1f[2] = {(float*)alloc((size_t)BATCH * N1 * 4), (float*)alloc((size_t)BATCH * N1 * 4)};
  bf16*  s0b[2] = {(bf16*)alloc((size_t)BATCH * N0 * 2), (bf16*)alloc((size_t)BATCH * N0 * 2)};
  bf16*  s1b[2] = {(bf16*)alloc((size_t)BATCH * N1 * 2), (bf16*)alloc((size_t)BATCH * N1 * 2)};

  auto blocks = [](int n) { return dim3((unsigned)((n + 255) / 256)); };
  // Per-launch precision conversion; weights are L2-resident afterwards.
  cvt_bf16_kernel<<<blocks(BATCH * NIN), 256, 0, stream>>>(data, rd, BATCH * NIN, 1);
  cvt_bf16_kernel<<<blocks(N0 * N1),     256, 0, stream>>>(w0, w0b, N0 * N1, 0);
  cvt_bf16_kernel<<<blocks(N1 * N0),     256, 0, stream>>>(w1, w1b, N1 * N0, 0);
  cvt_bf16_kernel<<<blocks(N1 * NIN),    256, 0, stream>>>(w2, w2b, N1 * NIN, 0);
  init_state_kernel<<<blocks(BATCH * N0), 256, 0, stream>>>(s0in, s0f[0], s0b[0], BATCH * N0);
  init_state_kernel<<<blocks(BATCH * N1), 256, 0, stream>>>(s1in, s1f[0], s1b[0], BATCH * N1);

  inp1_kernel<<<dim3(128), 128, 0, stream>>>(rd, w2b, b2, inp1);

  float* out_s0 = (float*)d_out;
  float* out_s1 = (float*)d_out + (size_t)BATCH * N0;
  for (int t = 0; t < TSTEPS; ++t) {
    const int p = t & 1, q = p ^ 1;
    float* o0 = (t == TSTEPS - 1) ? out_s0 : s0f[q];
    float* o1 = (t == TSTEPS - 1) ? out_s1 : s1f[q];
    step_kernel<<<dim3(160), 128, 0, stream>>>(
        s0f[p], s1f[p], s0b[p], s1b[p], w0b, w1b, b0, inp1,
        o0, o1, s0b[q], s1b[q]);
  }
}
